// MotionFormerDecoderLayer_68805376082089
// MI455X (gfx1250) — matrix operand order repaired and checked by reading, not compiled, not run on hardware
//
#include <hip/hip_runtime.h>

// ---------------- problem constants (match reference) ----------------
#define NHEADS   8
#define EMBED    256
#define DH       32
#define NPOINTS  4
#define HBEV     200
#define WBEV     200
#define HWBEV    (HBEV * WBEV)
#define FF       512
#define NB       4
#define NQ       1800
#define NAG      300
#define NMAP     100
#define OFFSCALE 1.0f

typedef __attribute__((ext_vector_type(16))) __bf16        v16bf;
typedef __attribute__((ext_vector_type(8)))  float         v8f;
typedef __attribute__((ext_vector_type(4)))  unsigned int  u32x4;
typedef __attribute__((ext_vector_type(2)))  unsigned int  u32x2;
typedef __attribute__((ext_vector_type(4)))  float         f32x4;

// float -> bf16 bits with round-to-nearest-even
__device__ __forceinline__ unsigned short bfbits(float f) {
    unsigned int u = __builtin_bit_cast(unsigned int, f);
    unsigned int r = u + 0x7FFFu + ((u >> 16) & 1u);
    return (unsigned short)(r >> 16);
}
__device__ __forceinline__ __bf16 f2bf(float f) {
    unsigned short h = bfbits(f);
    return __builtin_bit_cast(__bf16, h);
}
__device__ __forceinline__ float bf2f(__bf16 h) {
    unsigned int u = (unsigned int)__builtin_bit_cast(unsigned short, h) << 16;
    return __builtin_bit_cast(float, u);
}

union FragU { u32x4 q[2]; v16bf v; };

// =====================================================================
// Elementwise f32 -> bf16 (n must be a multiple of 4; all our tensors are)
// =====================================================================
__global__ __launch_bounds__(256)
void cvt_f32_bf16(const float* __restrict__ in, __bf16* __restrict__ out, long n4) {
    long i = (long)blockIdx.x * blockDim.x + threadIdx.x;
    if (i >= n4) return;
    f32x4 f = *(const f32x4*)(in + i * 4);
    unsigned int h01 = (unsigned int)bfbits(f.x) | ((unsigned int)bfbits(f.y) << 16);
    unsigned int h23 = (unsigned int)bfbits(f.z) | ((unsigned int)bfbits(f.w) << 16);
    u32x2 o = {h01, h23};
    *(u32x2*)(out + i * 4) = o;
}

// =====================================================================
// Weight convert + transpose: Wt[n*K + k] = bf16(W[k*N + n])  (col-major bf16)
// =====================================================================
__global__ __launch_bounds__(256)
void wtrans_kernel(const float* __restrict__ W, __bf16* __restrict__ Wt, int K, int N) {
    int idx = (int)(blockIdx.x * blockDim.x + threadIdx.x);
    if (idx >= K * N) return;
    int n = idx / K;
    int k = idx - n * K;
    Wt[idx] = f2bf(W[(size_t)k * N + n]);
}

// =====================================================================
// GEMM: C[M,N] = act(A[M,K] @ B[K,N] + bias[N])
// A: bf16 row-major (M x K). Bt: bf16 column-major (Bt[n*K + k]).
// Output f32 (Cf) or bf16 (Cbf, if non-null).
// One wave computes a 16 x (16*NT) strip; A fragment reused across NT tiles.
// K-loop is software-pipelined (double-buffered fragments) so the next
// k-step's global_load_b128s are in flight while the current WMMAs execute;
// the pre-WMMA s_wait then only covers the older load group instead of
// draining loadcnt to zero.
//
// Fragment layouts per CDNA5 ISA 7.12.2 (16-bit, wave32):
//   A lane {hi=lane>>4, m=lane&15}: v16bf elems 0..7  = A[m][kb+hi*8   .. +8)
//                                   elems 8..15       = A[m][kb+16+hi*8 .. +8)
//   B lane {hi, n=lane&15}:         v16bf elems 0..7  = B[kb+hi*8 .. +8)[n], elems 8..15 = B[kb+16+hi*8 .. +8)[n]
//   C/D: VGPR r -> row = r + hi*8, col = lane&15
// =====================================================================
template <int NT>
__global__ __launch_bounds__(128)
void gemm_bf16_wmma(const __bf16* __restrict__ A, const __bf16* __restrict__ Bt,
                    const float* __restrict__ bias, float* __restrict__ Cf,
                    __bf16* __restrict__ Cbf, int M, int N, int K, int relu) {
    int lane = threadIdx.x & 31;
    int wid  = (int)((blockIdx.x * blockDim.x + threadIdx.x) >> 5);
    int tilesN = N >> 4;
    int groups = tilesN / NT;          // tilesN % NT == 0 guaranteed by host
    int tilesM = M >> 4;
    if (wid >= tilesM * groups) return;
    int tm  = wid / groups;
    int tg  = wid - tm * groups;
    int tn0 = tg * NT;

    int m16 = lane & 15;
    int hi  = lane >> 4;

    const __bf16* Arow = A + (size_t)(tm * 16 + m16) * K;
    const __bf16* Bcol[NT];
#pragma unroll
    for (int t = 0; t < NT; ++t)
        Bcol[t] = Bt + (size_t)((tn0 + t) * 16 + m16) * K;

    v8f acc[NT];
#pragma unroll
    for (int t = 0; t < NT; ++t) acc[t] = {};

    // ---- prologue: fragments for kb = 0 ----
    FragU af;
    FragU bfr[NT];
    af.q[0] = *(const u32x4*)(Arow + hi * 8);
    af.q[1] = *(const u32x4*)(Arow + 16 + hi * 8);
#pragma unroll
    for (int t = 0; t < NT; ++t) {
        bfr[t].q[0] = *(const u32x4*)(Bcol[t] + hi * 8);
        bfr[t].q[1] = *(const u32x4*)(Bcol[t] + 16 + hi * 8);
    }

    // ---- pipelined main loop: prefetch k+1, compute k ----
    for (int kb = 32; kb < K; kb += 32) {
        FragU afn;
        FragU bfn[NT];
        afn.q[0] = *(const u32x4*)(Arow + kb + hi * 8);
        afn.q[1] = *(const u32x4*)(Arow + kb + 16 + hi * 8);
#pragma unroll
        for (int t = 0; t < NT; ++t) {
            bfn[t].q[0] = *(const u32x4*)(Bcol[t] + kb + hi * 8);
            bfn[t].q[1] = *(const u32x4*)(Bcol[t] + kb + 16 + hi * 8);
        }
#pragma unroll
        for (int t = 0; t < NT; ++t)
            acc[t] = __builtin_amdgcn_wmma_f32_16x16x32_bf16(
                false, af.v, false, bfr[t].v, (short)0, acc[t], false, false);
        af = afn;
#pragma unroll
        for (int t = 0; t < NT; ++t) bfr[t] = bfn[t];
    }

    // ---- epilogue: compute last k-step ----
#pragma unroll
    for (int t = 0; t < NT; ++t)
        acc[t] = __builtin_amdgcn_wmma_f32_16x16x32_bf16(
            false, af.v, false, bfr[t].v, (short)0, acc[t], false, false);

#pragma unroll
    for (int t = 0; t < NT; ++t) {
        int col = (tn0 + t) * 16 + m16;
        float bb = bias ? bias[col] : 0.0f;
#pragma unroll
        for (int r = 0; r < 8; ++r) {
            int row = tm * 16 + r + hi * 8;
            float v = acc[t][r] + bb;
            if (relu) v = fmaxf(v, 0.0f);
            if (Cbf) Cbf[(size_t)row * N + col] = f2bf(v);
            else     Cf [(size_t)row * N + col] = v;
        }
    }
}

// =====================================================================
// Flash-style MHA core: out[b,q,h*32+d] = softmax(q·k^T/sqrt(32)) @ v
// One wave per (b,h,q); each lane owns keys lane, lane+32, ...
// =====================================================================
__global__ __launch_bounds__(256)
void attn_kernel(const float* __restrict__ qp, const float* __restrict__ kp,
                 const float* __restrict__ vp, float* __restrict__ out,
                 int Lq, int Lk) {
    int lane = threadIdx.x & 31;
    int wid  = (int)((blockIdx.x * blockDim.x + threadIdx.x) >> 5);
    int total = NB * NHEADS * Lq;
    if (wid >= total) return;
    int q = wid % Lq;
    int t = wid / Lq;
    int h = t % NHEADS;
    int b = t / NHEADS;

    const float scale = 0.17677669529663687f; // 1/sqrt(32)

    const float* qrow = qp + ((size_t)b * Lq + q) * EMBED + h * DH;
    float qv[DH];
#pragma unroll
    for (int d = 0; d < DH; ++d) qv[d] = qrow[d];

    float m = -3.0e38f, l = 0.0f;
    float acc[DH];
#pragma unroll
    for (int d = 0; d < DH; ++d) acc[d] = 0.0f;

    const float* kbase = kp + (size_t)b * Lk * EMBED + h * DH;
    const float* vbase = vp + (size_t)b * Lk * EMBED + h * DH;

    for (int k0 = lane; k0 < Lk; k0 += 32) {
        const float* krow = kbase + (size_t)k0 * EMBED;
        float s = 0.0f;
#pragma unroll
        for (int d = 0; d < DH; ++d) s += qv[d] * krow[d];
        s *= scale;
        float mnew = fmaxf(m, s);
        float corr = __expf(m - mnew);
        float p    = __expf(s - mnew);
        l = l * corr + p;
        const float* vrow = vbase + (size_t)k0 * EMBED;
#pragma unroll
        for (int d = 0; d < DH; ++d) acc[d] = acc[d] * corr + p * vrow[d];
        m = mnew;
    }

    // butterfly merge across 32 lanes (every lane owns >=3 keys since Lk>=100)
#pragma unroll
    for (int off = 16; off >= 1; off >>= 1) {
        float m2 = __shfl_xor(m, off);
        float l2 = __shfl_xor(l, off);
        float mnew = fmaxf(m, m2);
        float c1 = __expf(m - mnew);
        float c2 = __expf(m2 - mnew);
        l = l * c1 + l2 * c2;
#pragma unroll
        for (int d = 0; d < DH; ++d) {
            float a2 = __shfl_xor(acc[d], off);
            acc[d] = acc[d] * c1 + a2 * c2;
        }
        m = mnew;
    }

    float inv = 1.0f / l;
    out[((size_t)b * Lq + q) * EMBED + h * DH + lane] = acc[lane] * inv;
}

// =====================================================================
// AddNorm: out = g * normalize(x + res) + b over last dim (256)
// =====================================================================
__global__ __launch_bounds__(256)
void addnorm_kernel(const float* __restrict__ x, const float* __restrict__ res,
                    const float* __restrict__ g, const float* __restrict__ bta,
                    float* __restrict__ out, int rows) {
    int lane = threadIdx.x & 31;
    int row  = (int)((blockIdx.x * blockDim.x + threadIdx.x) >> 5);
    if (row >= rows) return;
    const float* xr = x + (size_t)row * EMBED;
    const float* rr = res + (size_t)row * EMBED;
    float v[8];
    float s = 0.0f;
#pragma unroll
    for (int j = 0; j < 8; ++j) {
        int c = lane + 32 * j;
        v[j] = xr[c] + rr[c];
        s += v[j];
    }
#pragma unroll
    for (int off = 16; off >= 1; off >>= 1) s += __shfl_xor(s, off);
    float mu = s * (1.0f / EMBED);
    float s2 = 0.0f;
#pragma unroll
    for (int j = 0; j < 8; ++j) { float d = v[j] - mu; s2 += d * d; }
#pragma unroll
    for (int off = 16; off >= 1; off >>= 1) s2 += __shfl_xor(s2, off);
    float rstd = rsqrtf(s2 * (1.0f / EMBED) + 1e-5f);
    float* orow = out + (size_t)row * EMBED;
#pragma unroll
    for (int j = 0; j < 8; ++j) {
        int c = lane + 32 * j;
        orow[c] = g[c] * (v[j] - mu) * rstd + bta[c];
    }
}

// =====================================================================
// Deformable BEV sampling over bf16 value tensor (L2-resident: 82 MB).
// One wave per (b,q,h); lane = channel d.
// =====================================================================
__device__ __forceinline__ float bev_corner(const __bf16* __restrict__ vb, int x, int y) {
    if (x < 0 || x >= WBEV || y < 0 || y >= HBEV) return 0.0f;
    return bf2f(vb[(size_t)(y * WBEV + x) * EMBED]);
}

__global__ __launch_bounds__(256)
void deform_kernel(const __bf16* __restrict__ value, const float* __restrict__ off,
                   const float* __restrict__ alog, const float* __restrict__ ref,
                   float* __restrict__ out) {
    int lane = threadIdx.x & 31;
    int wid  = (int)((blockIdx.x * blockDim.x + threadIdx.x) >> 5);
    int total = NB * NQ * NHEADS;
    if (wid >= total) return;
    int h = wid % NHEADS;
    int t = wid / NHEADS;
    int q = t % NQ;
    int b = t / NQ;
    size_t bq = (size_t)b * NQ + q;

    float rx = ref[bq * 2 + 0];
    float ry = ref[bq * 2 + 1];

    float lg[NPOINTS];
    float mx = -3.0e38f;
#pragma unroll
    for (int p = 0; p < NPOINTS; ++p) {
        lg[p] = alog[bq * (NHEADS * NPOINTS) + h * NPOINTS + p];
        mx = fmaxf(mx, lg[p]);
    }
    float sum = 0.0f;
    float w[NPOINTS];
#pragma unroll
    for (int p = 0; p < NPOINTS; ++p) { w[p] = __expf(lg[p] - mx); sum += w[p]; }
    float inv = 1.0f / sum;

    const __bf16* vb = value + (size_t)b * HWBEV * EMBED + h * DH + lane;
    float acc = 0.0f;
#pragma unroll
    for (int p = 0; p < NPOINTS; ++p) {
        float ox = off[bq * (NHEADS * NPOINTS * 2) + h * (NPOINTS * 2) + p * 2 + 0];
        float oy = off[bq * (NHEADS * NPOINTS * 2) + h * (NPOINTS * 2) + p * 2 + 1];
        float locx = rx + ox * (OFFSCALE / (float)WBEV);
        float locy = ry + oy * (OFFSCALE / (float)HBEV);
        float xx = locx * WBEV - 0.5f;
        float yy = locy * HBEV - 0.5f;
        float x0f = floorf(xx), y0f = floorf(yy);
        float fx = xx - x0f, fy = yy - y0f;
        int x0 = (int)x0f, y0 = (int)y0f;
        float s = bev_corner(vb, x0,     y0    ) * (1.0f - fx) * (1.0f - fy)
                + bev_corner(vb, x0 + 1, y0    ) * fx          * (1.0f - fy)
                + bev_corner(vb, x0,     y0 + 1) * (1.0f - fx) * fy
                + bev_corner(vb, x0 + 1, y0 + 1) * fx          * fy;
        acc += (w[p] * inv) * s;
    }
    out[bq * EMBED + h * DH + lane] = acc;
}

// =====================================================================
// Concat [a | b | c] along last dim (256 each -> 768)
// =====================================================================
__global__ __launch_bounds__(256)
void concat3_kernel(const float* __restrict__ a, const float* __restrict__ b,
                    const float* __restrict__ c, float* __restrict__ out, int rows) {
    size_t idx = (size_t)blockIdx.x * blockDim.x + threadIdx.x;
    size_t total = (size_t)rows * (3 * EMBED);
    if (idx >= total) return;
    int col = (int)(idx % (3 * EMBED));
    size_t r = idx / (3 * EMBED);
    float v;
    if (col < EMBED)          v = a[r * EMBED + col];
    else if (col < 2 * EMBED) v = b[r * EMBED + (col - EMBED)];
    else                      v = c[r * EMBED + (col - 2 * EMBED)];
    out[idx] = v;
}

// =====================================================================
// Host orchestration
// =====================================================================
static inline void launch_gemm(const __bf16* A, const __bf16* Bt, const float* bias,
                               float* Cf, __bf16* Cbf, int M, int N, int K, int relu,
                               hipStream_t stream) {
    int tilesN = N >> 4;
    int tilesM = M >> 4;
    if ((tilesN & 3) == 0) {
        int tiles = tilesM * (tilesN >> 2);
        gemm_bf16_wmma<4><<<(tiles + 3) / 4, 128, 0, stream>>>(A, Bt, bias, Cf, Cbf, M, N, K, relu);
    } else {
        int tiles = tilesM * (tilesN >> 1);
        gemm_bf16_wmma<2><<<(tiles + 3) / 4, 128, 0, stream>>>(A, Bt, bias, Cf, Cbf, M, N, K, relu);
    }
}

static inline void launch_cvt(const float* in, __bf16* out, long n, hipStream_t stream) {
    long n4 = n >> 2;
    int blocks = (int)((n4 + 255) / 256);
    cvt_f32_bf16<<<blocks, 256, 0, stream>>>(in, out, n4);
}

static inline void launch_wtrans(const float* W, __bf16* Wt, int K, int N, hipStream_t stream) {
    int n = K * N;
    wtrans_kernel<<<(n + 255) / 256, 256, 0, stream>>>(W, Wt, K, N);
}

extern "C" void kernel_launch(void* const* d_in, const int* in_sizes, int n_in,
                              void* d_out, int out_size, void* d_ws, size_t ws_size,
                              hipStream_t stream) {
    const float* bev    = (const float*)d_in[0];
    const float* qry    = (const float*)d_in[1];
    const float* agent  = (const float*)d_in[2];
    const float* mapq   = (const float*)d_in[3];
    const float* refp   = (const float*)d_in[4];
    const float* saw    = (const float*)d_in[5];
    const float* sab    = (const float*)d_in[6];
    const float* caw    = (const float*)d_in[7];
    const float* cab    = (const float*)d_in[8];
    const float* lng    = (const float*)d_in[9];
    const float* lnb    = (const float*)d_in[10];
    const float* dwv    = (const float*)d_in[11];
    const float* dbv    = (const float*)d_in[12];
    const float* dwoff  = (const float*)d_in[13];
    const float* dboff  = (const float*)d_in[14];
    const float* dwattn = (const float*)d_in[15];
    const float* dbattn = (const float*)d_in[16];
    const float* dwo    = (const float*)d_in[17];
    const float* dbo    = (const float*)d_in[18];
    const float* w1     = (const float*)d_in[19];
    const float* b1     = (const float*)d_in[20];
    const float* w2     = (const float*)d_in[21];
    const float* b2     = (const float*)d_in[22];
    (void)in_sizes; (void)n_in; (void)out_size; (void)ws_size;

    const int ROWS = NB * NQ;                 // 7200
    const size_t RD = (size_t)ROWS * EMBED;   // 1.84M
    const int WMAT = EMBED * EMBED;           // 65536

    // ---------------- workspace carve-up ----------------
    float* ws = (float*)d_ws;
    size_t o = 0;
    float* qp   = ws + o; o += RD;
    float* kp   = ws + o; o += RD;
    float* vp   = ws + o; o += RD;
    float* tmp  = ws + o; o += RD;
    float* buf  = ws + o; o += RD;
    float* sa   = ws + o; o += RD;
    float* ag   = ws + o; o += RD;
    float* mp   = ws + o; o += RD;
    float* samp = ws + o; o += RD;
    float* gp   = ws + o; o += RD;
    float* offb = ws + o; o += (size_t)ROWS * 64;
    float* alog = ws + o; o += (size_t)ROWS * 32;
    float* ctx  = ws + o; o += (size_t)ROWS * 3 * EMBED;
    float* hid  = ws + o; o += (size_t)ROWS * FF;

    __bf16* bws = (__bf16*)(ws + o);
    size_t ob = 0;
    __bf16* qbf     = bws + ob; ob += RD;
    __bf16* sabf    = bws + ob; ob += RD;
    __bf16* tbf     = bws + ob; ob += RD;                           // attn-out / samp bf16
    __bf16* agbf    = bws + ob; ob += (size_t)NB * NAG * EMBED;
    __bf16* mapbf   = bws + ob; ob += (size_t)NB * NMAP * EMBED;
    __bf16* ctxbf   = bws + ob; ob += (size_t)ROWS * 3 * EMBED;
    __bf16* hidbf   = bws + ob; ob += (size_t)ROWS * FF;
    __bf16* bevbf   = bws + ob; ob += (size_t)NB * HWBEV * EMBED;   // 82 MB
    __bf16* valuebf = bws + ob; ob += (size_t)NB * HWBEV * EMBED;   // 82 MB
    // transposed bf16 weights
    __bf16* wt_sa[4]; __bf16* wt_ca[4];
    for (int i = 0; i < 4; ++i) { wt_sa[i] = bws + ob; ob += WMAT; }
    for (int i = 0; i < 4; ++i) { wt_ca[i] = bws + ob; ob += WMAT; }
    __bf16* wt_dwv   = bws + ob; ob += WMAT;
    __bf16* wt_dwoff = bws + ob; ob += EMBED * 64;
    __bf16* wt_dwatt = bws + ob; ob += EMBED * 32;
    __bf16* wt_dwo   = bws + ob; ob += WMAT;
    __bf16* wt_w1    = bws + ob; ob += 3 * EMBED * FF;
    __bf16* wt_w2    = bws + ob; ob += FF * EMBED;

    // ---------------- one-time conversions (per launch) ----------------
    launch_cvt(qry,   qbf,   (long)RD, stream);
    launch_cvt(agent, agbf,  (long)NB * NAG * EMBED, stream);
    launch_cvt(mapq,  mapbf, (long)NB * NMAP * EMBED, stream);
    launch_cvt(bev,   bevbf, (long)NB * HWBEV * EMBED, stream);
    for (int i = 0; i < 4; ++i) launch_wtrans(saw + i * WMAT, wt_sa[i], EMBED, EMBED, stream);
    for (int i = 0; i < 4; ++i) launch_wtrans(caw + i * WMAT, wt_ca[i], EMBED, EMBED, stream);
    launch_wtrans(dwv,    wt_dwv,   EMBED, EMBED, stream);
    launch_wtrans(dwoff,  wt_dwoff, EMBED, 64,    stream);
    launch_wtrans(dwattn, wt_dwatt, EMBED, 32,    stream);
    launch_wtrans(dwo,    wt_dwo,   EMBED, EMBED, stream);
    launch_wtrans(w1,     wt_w1,    3 * EMBED, FF, stream);
    launch_wtrans(w2,     wt_w2,    FF, EMBED,    stream);

    const int attnWaves = NB * NHEADS * NQ;
    const int attnBlocks = (attnWaves * 32 + 255) / 256;
    const int rowBlocks  = (ROWS * 32 + 255) / 256;

    // ---------------- self attention ----------------
    launch_gemm(qbf, wt_sa[0], sab + 0 * EMBED, qp, nullptr, ROWS, EMBED, EMBED, 0, stream);
    launch_gemm(qbf, wt_sa[1], sab + 1 * EMBED, kp, nullptr, ROWS, EMBED, EMBED, 0, stream);
    launch_gemm(qbf, wt_sa[2], sab + 2 * EMBED, vp, nullptr, ROWS, EMBED, EMBED, 0, stream);
    attn_kernel<<<attnBlocks, 256, 0, stream>>>(qp, kp, vp, tmp, NQ, NQ);
    launch_cvt(tmp, tbf, (long)RD, stream);
    launch_gemm(tbf, wt_sa[3], sab + 3 * EMBED, buf, nullptr, ROWS, EMBED, EMBED, 0, stream);
    addnorm_kernel<<<rowBlocks, 256, 0, stream>>>(buf, qry, lng + 0 * EMBED, lnb + 0 * EMBED, sa, ROWS);
    launch_cvt(sa, sabf, (long)RD, stream);

    // ---------------- agent cross attention ----------------
    launch_gemm(sabf, wt_ca[0], cab + 0 * EMBED, qp, nullptr, ROWS, EMBED, EMBED, 0, stream);
    launch_gemm(agbf, wt_ca[1], cab + 1 * EMBED, kp, nullptr, NB * NAG, EMBED, EMBED, 0, stream);
    launch_gemm(agbf, wt_ca[2], cab + 2 * EMBED, vp, nullptr, NB * NAG, EMBED, EMBED, 0, stream);
    attn_kernel<<<attnBlocks, 256, 0, stream>>>(qp, kp, vp, tmp, NQ, NAG);
    launch_cvt(tmp, tbf, (long)RD, stream);
    launch_gemm(tbf, wt_ca[3], cab + 3 * EMBED, buf, nullptr, ROWS, EMBED, EMBED, 0, stream);
    addnorm_kernel<<<rowBlocks, 256, 0, stream>>>(buf, sa, lng + 1 * EMBED, lnb + 1 * EMBED, ag, ROWS);

    // ---------------- map cross attention (Q projection of sa reused in qp) ----------------
    launch_gemm(mapbf, wt_ca[1], cab + 1 * EMBED, kp, nullptr, NB * NMAP, EMBED, EMBED, 0, stream);
    launch_gemm(mapbf, wt_ca[2], cab + 2 * EMBED, vp, nullptr, NB * NMAP, EMBED, EMBED, 0, stream);
    attn_kernel<<<attnBlocks, 256, 0, stream>>>(qp, kp, vp, tmp, NQ, NMAP);
    launch_cvt(tmp, tbf, (long)RD, stream);
    launch_gemm(tbf, wt_ca[3], cab + 3 * EMBED, buf, nullptr, ROWS, EMBED, EMBED, 0, stream);
    addnorm_kernel<<<rowBlocks, 256, 0, stream>>>(buf, sa, lng + 2 * EMBED, lnb + 2 * EMBED, mp, ROWS);

    // ---------------- deformable BEV attention ----------------
    launch_gemm(bevbf, wt_dwv, dbv, nullptr, valuebf, NB * HWBEV, EMBED, EMBED, 0, stream);
    launch_gemm(qbf, wt_dwoff, dboff,  offb, nullptr, ROWS, NHEADS * NPOINTS * 2, EMBED, 0, stream);
    launch_gemm(qbf, wt_dwatt, dbattn, alog, nullptr, ROWS, NHEADS * NPOINTS,     EMBED, 0, stream);
    {
        int waves = NB * NQ * NHEADS;
        int blocks = (waves * 32 + 255) / 256;
        deform_kernel<<<blocks, 256, 0, stream>>>(valuebf, offb, alog, refp, samp);
    }
    launch_cvt(samp, tbf, (long)RD, stream);
    launch_gemm(tbf, wt_dwo, dbo, gp, nullptr, ROWS, EMBED, EMBED, 0, stream);

    // ---------------- fuse + MLP ----------------
    {
        size_t total = (size_t)ROWS * 3 * EMBED;
        int blocks = (int)((total + 255) / 256);
        concat3_kernel<<<blocks, 256, 0, stream>>>(ag, mp, gp, ctx, ROWS);
    }
    launch_cvt(ctx, ctxbf, (long)ROWS * 3 * EMBED, stream);
    launch_gemm(ctxbf, wt_w1, b1, hid, nullptr, ROWS, FF, 3 * EMBED, 1, stream);
    launch_cvt(hid, hidbf, (long)ROWS * FF, stream);
    launch_gemm(hidbf, wt_w2, b2, (float*)d_out, nullptr, ROWS, EMBED, FF, 0, stream);
}
